// Net_58291296141394
// MI455X (gfx1250) — matrix operand order repaired and checked by reading, not compiled, not run on hardware
//
#include <hip/hip_runtime.h>

// RNN sizes (compile-time constants from the reference)
#define BB     256
#define TT     512
#define NN     256
#define NIN    6
#define ALPHA_ 0.2f
#define NOISE_STD_ 0.09486832980505137f   // sqrt(2*0.2*0.15^2)

#define MT   16      // batch rows per workgroup (WMMA M)
#define ROWP 264     // state panel row stride in bf16 elements (256 + 8 pad)
#define ROWU 40      // u panel row stride in bf16 elements (32 + 8 pad)

typedef __attribute__((ext_vector_type(16))) __bf16 v16bf;
typedef __attribute__((ext_vector_type(8)))  __bf16 v8bf;
typedef __attribute__((ext_vector_type(8)))  float  v8f;

// Raw split workgroup barrier: only require this wave's LDS ops complete
// (DScnt==0) before signaling -- do NOT drain global stores/loads like
// __syncthreads() does.
#define BARRIER_SIGNAL() \
    asm volatile("s_wait_dscnt 0x0\n\ts_barrier_signal -1" ::: "memory")
#define BARRIER_WAIT() \
    asm volatile("s_barrier_wait -1" ::: "memory")

__global__ __launch_bounds__(512, 1)
void rnn_leaky_kernel(const float* __restrict__ u,
                      const float* __restrict__ noise,
                      const float* __restrict__ Wrec,
                      const float* __restrict__ Win,
                      float* __restrict__ out)
{
    // Double-buffered bf16 state panel [16 rows][256 cols] + augmented
    // u panel [16 rows][32 cols] (only K=0..5 live, rest stays zero).
    __shared__ __align__(16) __bf16 sA [2][MT * ROWP];
    __shared__ __align__(16) __bf16 sAU[2][MT * ROWU];

    const int tid  = threadIdx.x;
    const int lane = tid & 31;
    const int wave = tid >> 5;        // 16 waves -> 16 column tiles of N
    const int hf   = lane >> 4;       // half-wave select (0/1)
    const int ln   = lane & 15;
    const int cn   = wave * 16 + ln;  // this lane's output column n (C/D layout)
    const int b0   = blockIdx.x * MT; // batch tile base

    // Zero-fill both u panels once; the K=6..31 padding lanes stay zero forever.
    for (int j = tid; j < 2 * MT * ROWU; j += 512)
        (&sAU[0][0])[j] = (__bf16)0.0f;

    // B fragments: W_rec^T column tile, K=256 split into 8 fragments of 32.
    // B[k][n] = W_rec[n][k] -> row n of W_rec, contiguous over k.
    // 16-bit B layout: lane half hf holds K=8hf..8hf+7 (e<8) and K=16+8hf..16+8hf+7 (e>=8),
    // matching the A fragment order.
    v16bf bfrag[8];
    {
        const float* wrow = Wrec + (size_t)cn * NN;
#pragma unroll
        for (int kf = 0; kf < 8; ++kf) {
            const int k0 = kf * 32 + hf * 8;
#pragma unroll
            for (int e = 0; e < 16; ++e)
                bfrag[kf][e] = (__bf16)wrow[k0 + e + (e >= 8 ? 8 : 0)];
        }
    }

    // B fragment for the augmented input block: B[k][n] = W_in[n][k], K=0..31,
    // only K<6 nonzero. Built once, lives in 8 VGPRs.
    v16bf bwin;
#pragma unroll
    for (int e = 0; e < 16; ++e) {
        const int k = hf * 8 + e + (e >= 8 ? 8 : 0);
        bwin[e] = (k < NIN) ? (__bf16)Win[cn * NIN + k] : (__bf16)0.0f;
    }

    // Persistent f32 state tile in WMMA C/D layout: s[r] = state[m][cn], m = r + 8*hf.
    v8f s = {};

    // t = 0 output is the zero initial state.
#pragma unroll
    for (int r = 0; r < 8; ++r) {
        const int m = r + 8 * hf;
        out[((size_t)(b0 + m) * TT + 0) * NN + cn] = 0.0f;
    }

    int buf = 0;
    for (int t = 0; t < TT - 1; ++t) {
        // 1) issue this step's noise loads EARLY: their latency overlaps the
        //    LDS publishes and the barrier below. Also prefetch step t+1.
        float nz[8];
#pragma unroll
        for (int r = 0; r < 8; ++r) {
            const int m = r + 8 * hf;
            const size_t nz_idx = ((size_t)(b0 + m) * TT + t) * NN + cn;
            nz[r] = noise[nz_idx];
            __builtin_prefetch(&noise[nz_idx + NN], 0, 0);  // global_prefetch_b8
        }

        // 2) publish bf16 copy of the state tile into the LDS A panel
#pragma unroll
        for (int r = 0; r < 8; ++r) {
            const int m = r + 8 * hf;
            sA[buf][m * ROWP + cn] = (__bf16)s[r];
        }
        // 3) stage u[b0:b0+16, t, 0:6] as bf16 into the augmented A panel
        if (tid < MT * NIN) {
            const int m = tid / NIN, i = tid % NIN;
            sAU[buf][m * ROWU + i] =
                (__bf16)u[((size_t)(b0 + m) * TT + t) * NIN + i];
        }
        BARRIER_SIGNAL();   // s_wait_dscnt 0 ; s_barrier_signal -1

        // 4) accumulator seed: sigma * noise. The s_wait_loadcnt for nz lands
        //    here, overlapping the other waves' progress to the barrier.
        v8f c;
#pragma unroll
        for (int r = 0; r < 8; ++r) c[r] = NOISE_STD_ * nz[r];

        BARRIER_WAIT();     // s_barrier_wait -1

        // 5) preload ALL A fragments (burst of ds_load_b128), then run the
        //    9-deep WMMA chain; LDS latency hides under earlier WMMAs.
        //    A 16x32 layout per lane: row ln, K = k0..k0+7 (V0-3) and
        //    K = k0+16..k0+23 (V4-7), k0 = 8*hf.
        v16bf au;
        {
            const __bf16* rowu = &sAU[buf][ln * ROWU];
            v8bf alo = *(const v8bf*)(rowu + hf * 8);
            v8bf ahi = *(const v8bf*)(rowu + hf * 8 + 16);
#pragma unroll
            for (int e = 0; e < 8; ++e) { au[e] = alo[e]; au[e + 8] = ahi[e]; }
        }
        v16bf afr[8];
        {
            const __bf16* arow = &sA[buf][ln * ROWP];
#pragma unroll
            for (int kf = 0; kf < 8; ++kf) {
                const int k0 = kf * 32 + hf * 8;
                v8bf alo = *(const v8bf*)(arow + k0);        // K = k0 .. k0+7
                v8bf ahi = *(const v8bf*)(arow + k0 + 16);   // K = k0+16 .. k0+23
#pragma unroll
                for (int e = 0; e < 8; ++e) { afr[kf][e] = alo[e]; afr[kf][e + 8] = ahi[e]; }
            }
        }

        c = __builtin_amdgcn_wmma_f32_16x16x32_bf16(
                false, au, false, bwin, (short)0, c, false, false);
#pragma unroll
        for (int kf = 0; kf < 8; ++kf)
            c = __builtin_amdgcn_wmma_f32_16x16x32_bf16(
                    false, afr[kf], false, bfrag[kf], (short)0, c, false, false);

        // 6) relu + leaky blend in f32, emit state t+1
#pragma unroll
        for (int r = 0; r < 8; ++r) {
            const float rl = __builtin_fmaxf(c[r], 0.0f);   // single v_max_num_f32
            s[r] = s[r] + ALPHA_ * (rl - s[r]);             // (1-a)s + a*relu
            const int m = r + 8 * hf;
            out[((size_t)(b0 + m) * TT + (size_t)(t + 1)) * NN + cn] = s[r];
        }

        buf ^= 1;   // double buffer -> one barrier per step is sufficient
    }
}

extern "C" void kernel_launch(void* const* d_in, const int* in_sizes, int n_in,
                              void* d_out, int out_size, void* d_ws, size_t ws_size,
                              hipStream_t stream) {
    (void)in_sizes; (void)n_in; (void)d_ws; (void)ws_size; (void)out_size;
    const float* u     = (const float*)d_in[0];  // [B, T, IN]
    const float* noise = (const float*)d_in[1];  // [B, T, N]
    const float* Wrec  = (const float*)d_in[2];  // [N, N]
    const float* Win   = (const float*)d_in[3];  // [N, IN]
    float* out = (float*)d_out;                  // [B, T, N]

    dim3 grid(BB / MT);   // 16 workgroups, one batch tile each
    dim3 block(512);      // 16 waves: one 16-column N tile per wave
    rnn_leaky_kernel<<<grid, block, 0, stream>>>(u, noise, Wrec, Win, out);
}
